// GraphSAGEClassifier_73229192397463
// MI455X (gfx1250) — compile-verified, matrix operand order checked by reading, not run on hardware
//
#include <hip/hip_runtime.h>

typedef __attribute__((ext_vector_type(16))) _Float16 v16h;
typedef __attribute__((ext_vector_type(8)))  _Float16 v8h;
typedef __attribute__((ext_vector_type(8)))  float    v8f;

#define N_NODES  50000
#define N_EDGES  200000
#define IN_DIM   384
#define HID      128
#define OUT_DIM  64
#define N_CLS    2
#define N_GRAPHS 64

// number of 16-wide N tiles each wave computes (16 x 64 output strip per wave)
#define NT 4

// ---------------------------------------------------------------- utilities
__global__ void fill_zero_f32(float* __restrict__ p, int n) {
    int i = blockIdx.x * blockDim.x + threadIdx.x;
    if (i < n) p[i] = 0.0f;
}

__global__ void degree_count(const int* __restrict__ dst, float* __restrict__ cnt) {
    int e = blockIdx.x * blockDim.x + threadIdx.x;
    if (e < N_EDGES) unsafeAtomicAdd(&cnt[dst[e]], 1.0f);
}

// scatter-add feat[src] into agg[dst]; one thread handles 4 floats of one edge
__global__ void scatter_add(const float* __restrict__ feat, const int* __restrict__ src,
                            const int* __restrict__ dst, float* __restrict__ agg, int D4) {
    int t = blockIdx.x * blockDim.x + threadIdx.x;
    if (t >= N_EDGES * D4) return;
    int e = t / D4, c = t % D4;
    int s = src[e], d = dst[e];
    const float4 v = *((const float4*)(feat + (long)s * D4 * 4) + c);
    float* a = agg + (long)d * D4 * 4 + c * 4;
    unsafeAtomicAdd(a + 0, v.x);
    unsafeAtomicAdd(a + 1, v.y);
    unsafeAtomicAdd(a + 2, v.z);
    unsafeAtomicAdd(a + 3, v.w);
}

// W [K x N] fp32 row-major -> Wt [N x K] f16 (column-major view for B fragments)
__global__ void wcvt_t(const float* __restrict__ W, _Float16* __restrict__ Wt, int K, int N) {
    int t = blockIdx.x * blockDim.x + threadIdx.x;
    if (t >= K * N) return;
    int k = t / N, n = t % N;
    Wt[(long)n * K + k] = (_Float16)W[(long)k * N + n];
}

// ---------------------------------------------------------------- fused SAGE GEMM
// out[M,N] = relu( (agg/cnt) @ Wl + x @ Wr + bias )
// one wave32 computes a 16 x (16*NT) output strip: A fragments (agg & x rows,
// converted f32->f16 with the mean normalization folded in) are built ONCE per
// K-step and reused across NT column tiles -> 2*NT WMMAs per K-step.
__global__ void sage_gemm(const float* __restrict__ agg, const float* __restrict__ cnt,
                          const float* __restrict__ xin,
                          const _Float16* __restrict__ WlT, const _Float16* __restrict__ WrT,
                          const float* __restrict__ bias, float* __restrict__ out,
                          int M, int K, int N) {
    const int stripsN = N / (16 * NT);
    const int strips  = (M >> 4) * stripsN;
    const int gw = blockIdx.x * (blockDim.x >> 5) + (threadIdx.x >> 5);
    if (gw >= strips) return;
    const int lane = threadIdx.x & 31;
    const int m0 = (gw / stripsN) << 4;
    const int n0 = (gw % stripsN) * (16 * NT);
    const int row  = m0 + (lane & 15);         // A fragment row (both lane halves)
    const int col0 = n0 + (lane & 15);         // B/D column of tile 0
    const int klo  = (lane >> 4) << 3;         // 0 for lanes 0-15, 8 for lanes 16-31

    const float invc = 1.0f / fmaxf(cnt[row], 1.0f);
    const float* __restrict__ aggRow = agg + (long)row * K;
    const float* __restrict__ xRow   = xin + (long)row * K;
    const _Float16* __restrict__ wl  = WlT + (long)col0 * K;
    const _Float16* __restrict__ wr  = WrT + (long)col0 * K;

    v8f acc[NT];
#pragma unroll
    for (int t = 0; t < NT; ++t) acc[t] = (v8f){};

    for (int kt = 0; kt < K; kt += 32) {
        // ---- build shared A fragments (agg/cnt and x) once per K-step
        v16h aA, aX;
#pragma unroll
        for (int i = 0; i < 8; ++i) {
            const int k0 = kt + klo + i;        // K sub-block 0..7 (or 8..15)
            const int k1 = kt + 16 + klo + i;   // K sub-block 16..23 (or 24..31)
            aA[i]     = (_Float16)(aggRow[k0] * invc);
            aA[i + 8] = (_Float16)(aggRow[k1] * invc);
            aX[i]     = (_Float16)xRow[k0];
            aX[i + 8] = (_Float16)xRow[k1];
        }
        // ---- per column tile: load B fragments, 2 WMMAs each
#pragma unroll
        for (int t = 0; t < NT; ++t) {
            const long bo = (long)t * 16 * K + kt + klo;
            const v8h bl0 = *(const v8h*)(wl + bo);
            const v8h bl1 = *(const v8h*)(wl + bo + 16);
            const v8h br0 = *(const v8h*)(wr + bo);
            const v8h br1 = *(const v8h*)(wr + bo + 16);
            v16h bL, bR;
#pragma unroll
            for (int i = 0; i < 8; ++i) {
                bL[i] = bl0[i]; bL[i + 8] = bl1[i];
                bR[i] = br0[i]; bR[i + 8] = br1[i];
            }
            acc[t] = __builtin_amdgcn_wmma_f32_16x16x32_f16(false, aA, false, bL,
                                                            (short)0, acc[t], false, false);
            acc[t] = __builtin_amdgcn_wmma_f32_16x16x32_f16(false, aX, false, bR,
                                                            (short)0, acc[t], false, false);
        }
    }

    const int rbase = m0 + ((lane >> 4) << 3); // D: VGPR r -> M = r (lanes<16) / r+8
#pragma unroll
    for (int t = 0; t < NT; ++t) {
        const int col = col0 + t * 16;
        const float b = bias[col];
#pragma unroll
        for (int r = 0; r < 8; ++r) {
            out[(long)(rbase + r) * N + col] = fmaxf(acc[t][r] + b, 0.0f);
        }
    }
}

// ---------------------------------------------------------------- pooling + classifier
__global__ void pool_scatter(const float* __restrict__ h, const int* __restrict__ batch,
                             float* __restrict__ pool, float* __restrict__ gcnt) {
    int t = blockIdx.x * blockDim.x + threadIdx.x;
    if (t >= N_NODES * OUT_DIM) return;
    int i = t / OUT_DIM, d = t % OUT_DIM;
    int g = batch[i];
    unsafeAtomicAdd(&pool[g * OUT_DIM + d], h[t]);
    if (d == 0) unsafeAtomicAdd(&gcnt[g], 1.0f);
}

__global__ void classifier(const float* __restrict__ pool, const float* __restrict__ gcnt,
                           const float* __restrict__ Wc, const float* __restrict__ bc,
                           float* __restrict__ out) {
    int t = blockIdx.x * blockDim.x + threadIdx.x;
    if (t >= N_GRAPHS * N_CLS) return;
    int g = t / N_CLS, c = t % N_CLS;
    float invc = 1.0f / fmaxf(gcnt[g], 1.0f);
    float s = bc[c];
    for (int k = 0; k < OUT_DIM; ++k)
        s += pool[g * OUT_DIM + k] * invc * Wc[k * N_CLS + c];
    out[t] = s;
}

// ---------------------------------------------------------------- launch
extern "C" void kernel_launch(void* const* d_in, const int* in_sizes, int n_in,
                              void* d_out, int out_size, void* d_ws, size_t ws_size,
                              hipStream_t stream) {
    const float* x    = (const float*)d_in[0];
    const int*   ei   = (const int*)d_in[1];     // [2, N_EDGES]
    const int*   batch= (const int*)d_in[2];
    const float* W_l1 = (const float*)d_in[3];
    const float* b1   = (const float*)d_in[4];
    const float* W_r1 = (const float*)d_in[5];
    const float* W_l2 = (const float*)d_in[6];
    const float* b2   = (const float*)d_in[7];
    const float* W_r2 = (const float*)d_in[8];
    const float* W_l3 = (const float*)d_in[9];
    const float* b3   = (const float*)d_in[10];
    const float* W_r3 = (const float*)d_in[11];
    const float* Wc   = (const float*)d_in[12];
    const float* bc   = (const float*)d_in[13];
    const int* src = ei;
    const int* dst = ei + N_EDGES;

    // workspace carve-out (256B aligned)
    char*  ws  = (char*)d_ws;
    size_t off = 0;
    auto carve = [&](size_t bytes) -> void* {
        void* p = ws + off;
        off = (off + bytes + 255) & ~(size_t)255;
        return p;
    };
    float*     agg   = (float*)carve((size_t)N_NODES * IN_DIM * 4);
    float*     cnt   = (float*)carve((size_t)N_NODES * 4);
    float*     hA    = (float*)carve((size_t)N_NODES * HID * 4);
    float*     hB    = (float*)carve((size_t)N_NODES * HID * 4);
    _Float16*  WlT1  = (_Float16*)carve((size_t)IN_DIM * HID * 2);
    _Float16*  WrT1  = (_Float16*)carve((size_t)IN_DIM * HID * 2);
    _Float16*  WlT2  = (_Float16*)carve((size_t)HID * HID * 2);
    _Float16*  WrT2  = (_Float16*)carve((size_t)HID * HID * 2);
    _Float16*  WlT3  = (_Float16*)carve((size_t)HID * OUT_DIM * 2);
    _Float16*  WrT3  = (_Float16*)carve((size_t)HID * OUT_DIM * 2);
    float*     pool  = (float*)carve((size_t)(N_GRAPHS * OUT_DIM + N_GRAPHS) * 4);
    float*     gcnt  = pool + N_GRAPHS * OUT_DIM;

    const int T = 256;
    auto nb = [](long n, int t) { return (int)((n + t - 1) / t); };

    // degree counts (shared by all 3 layers)
    fill_zero_f32<<<nb(N_NODES, T), T, 0, stream>>>(cnt, N_NODES);
    degree_count<<<nb(N_EDGES, T), T, 0, stream>>>(dst, cnt);

    // weights -> f16, transposed
    wcvt_t<<<nb((long)IN_DIM * HID, T), T, 0, stream>>>(W_l1, WlT1, IN_DIM, HID);
    wcvt_t<<<nb((long)IN_DIM * HID, T), T, 0, stream>>>(W_r1, WrT1, IN_DIM, HID);
    wcvt_t<<<nb((long)HID * HID, T), T, 0, stream>>>(W_l2, WlT2, HID, HID);
    wcvt_t<<<nb((long)HID * HID, T), T, 0, stream>>>(W_r2, WrT2, HID, HID);
    wcvt_t<<<nb((long)HID * OUT_DIM, T), T, 0, stream>>>(W_l3, WlT3, HID, OUT_DIM);
    wcvt_t<<<nb((long)HID * OUT_DIM, T), T, 0, stream>>>(W_r3, WrT3, HID, OUT_DIM);

    // ---- layer 1: x[50000,384] -> hA[50000,128]
    {
        const int K = IN_DIM, N = HID, D4 = K / 4;
        fill_zero_f32<<<nb((long)N_NODES * K, T), T, 0, stream>>>(agg, N_NODES * K);
        scatter_add<<<nb((long)N_EDGES * D4, T), T, 0, stream>>>(x, src, dst, agg, D4);
        int strips = (N_NODES / 16) * (N / (16 * NT));
        sage_gemm<<<nb(strips, 8), T, 0, stream>>>(agg, cnt, x, WlT1, WrT1, b1, hA,
                                                   N_NODES, K, N);
    }
    // ---- layer 2: hA[50000,128] -> hB[50000,128]
    {
        const int K = HID, N = HID, D4 = K / 4;
        fill_zero_f32<<<nb((long)N_NODES * K, T), T, 0, stream>>>(agg, N_NODES * K);
        scatter_add<<<nb((long)N_EDGES * D4, T), T, 0, stream>>>(hA, src, dst, agg, D4);
        int strips = (N_NODES / 16) * (N / (16 * NT));
        sage_gemm<<<nb(strips, 8), T, 0, stream>>>(agg, cnt, hA, WlT2, WrT2, b2, hB,
                                                   N_NODES, K, N);
    }
    // ---- layer 3: hB[50000,128] -> hA[50000,64] (reuse)
    {
        const int K = HID, N = OUT_DIM, D4 = K / 4;
        fill_zero_f32<<<nb((long)N_NODES * K, T), T, 0, stream>>>(agg, N_NODES * K);
        scatter_add<<<nb((long)N_EDGES * D4, T), T, 0, stream>>>(hB, src, dst, agg, D4);
        int strips = (N_NODES / 16) * (N / (16 * NT));
        sage_gemm<<<nb(strips, 8), T, 0, stream>>>(agg, cnt, hB, WlT3, WrT3, b3, hA,
                                                   N_NODES, K, N);
    }

    // ---- global mean pool + classifier -> d_out [64,2]
    fill_zero_f32<<<nb(N_GRAPHS * OUT_DIM + N_GRAPHS, T), T, 0, stream>>>(
        pool, N_GRAPHS * OUT_DIM + N_GRAPHS);
    pool_scatter<<<nb((long)N_NODES * OUT_DIM, T), T, 0, stream>>>(hA, batch, pool, gcnt);
    classifier<<<1, 128, 0, stream>>>(pool, gcnt, Wc, bc, (float*)d_out);
}
